// RWKV_Predictor_65816078844387
// MI455X (gfx1250) — matrix operand order repaired and checked by reading, not compiled, not run on hardware
//
#include <hip/hip_runtime.h>
#include <hip/hip_bf16.h>

typedef __attribute__((ext_vector_type(16))) _Float16 v16h;
typedef __attribute__((ext_vector_type(8)))  _Float16 v8h;
typedef __attribute__((ext_vector_type(8)))  float    v8f;

static constexpr int Bq = 16;
static constexpr int Tq = 1024;
static constexpr int Dq = 512;
static constexpr int Lq = 6;
static constexpr int Fq = 1024;
static constexpr int Mq = Bq * Tq;   // 16384 rows

static __device__ __forceinline__ float sigf(float x) {
    return 1.0f / (1.0f + __expf(-x));
}

// ---------------------------------------------------------------------------
// f32 -> f16 conversion (weights), grid-stride
// ---------------------------------------------------------------------------
__global__ void k_f32_to_f16(const float* __restrict__ s, _Float16* __restrict__ d, size_t n) {
    size_t i = (size_t)blockIdx.x * blockDim.x + threadIdx.x;
    size_t stride = (size_t)gridDim.x * blockDim.x;
    for (; i < n; i += stride) d[i] = (_Float16)s[i];
}

// ---------------------------------------------------------------------------
// Embedding gather + LayerNorm (one 256-thread block per row of D=512)
// ---------------------------------------------------------------------------
__global__ __launch_bounds__(256) void k_embed_ln(const int* __restrict__ y,
        const float* __restrict__ emb, const float* __restrict__ w,
        const float* __restrict__ b, float* __restrict__ out) {
    int row = blockIdx.x;
    int tok = y[row];
    const float* src = emb + (size_t)tok * Dq;
    int d0 = threadIdx.x, d1 = threadIdx.x + 256;
    float a0 = src[d0], a1 = src[d1];
    float s = a0 + a1, q = a0 * a0 + a1 * a1;
    __shared__ float sm[16];
    for (int o = 16; o > 0; o >>= 1) {
        s += __shfl_down(s, o, 32);
        q += __shfl_down(q, o, 32);
    }
    int lane = threadIdx.x & 31, wid = threadIdx.x >> 5;
    if (lane == 0) { sm[wid] = s; sm[wid + 8] = q; }
    __syncthreads();
    if (threadIdx.x == 0) {
        float ts = 0.f, tq = 0.f;
        for (int i = 0; i < 8; i++) { ts += sm[i]; tq += sm[i + 8]; }
        sm[0] = ts; sm[8] = tq;
    }
    __syncthreads();
    float m   = sm[0] * (1.0f / Dq);
    float var = sm[8] * (1.0f / Dq) - m * m;
    float inv = rsqrtf(var + 1e-5f);
    size_t base = (size_t)row * Dq;
    out[base + d0] = (a0 - m) * inv * w[d0] + b[d0];
    out[base + d1] = (a1 - m) * inv * w[d1] + b[d1];
}

// ---------------------------------------------------------------------------
// LayerNorm of x -> f16 (for LN2 / GEMM input), one block per row
// ---------------------------------------------------------------------------
__global__ __launch_bounds__(256) void k_ln_f16(const float* __restrict__ x,
        const float* __restrict__ w, const float* __restrict__ b,
        _Float16* __restrict__ out) {
    int row = blockIdx.x;
    const float* src = x + (size_t)row * Dq;
    int d0 = threadIdx.x, d1 = threadIdx.x + 256;
    float a0 = src[d0], a1 = src[d1];
    float s = a0 + a1, q = a0 * a0 + a1 * a1;
    __shared__ float sm[16];
    for (int o = 16; o > 0; o >>= 1) {
        s += __shfl_down(s, o, 32);
        q += __shfl_down(q, o, 32);
    }
    int lane = threadIdx.x & 31, wid = threadIdx.x >> 5;
    if (lane == 0) { sm[wid] = s; sm[wid + 8] = q; }
    __syncthreads();
    if (threadIdx.x == 0) {
        float ts = 0.f, tq = 0.f;
        for (int i = 0; i < 8; i++) { ts += sm[i]; tq += sm[i + 8]; }
        sm[0] = ts; sm[8] = tq;
    }
    __syncthreads();
    float m   = sm[0] * (1.0f / Dq);
    float var = sm[8] * (1.0f / Dq) - m * m;
    float inv = rsqrtf(var + 1e-5f);
    size_t base = (size_t)row * Dq;
    out[base + d0] = (_Float16)((a0 - m) * inv * w[d0] + b[d0]);
    out[base + d1] = (_Float16)((a1 - m) * inv * w[d1] + b[d1]);
}

// ---------------------------------------------------------------------------
// Fused LN1 + token shift: xm[t] = 0.5*(LN(x[t]) + LN(x[t-1])), f16 out.
// Recomputes the previous row's LN stats (cheap VALU) to avoid a 64MB/layer
// f32 round-trip through memory.
// ---------------------------------------------------------------------------
__global__ __launch_bounds__(256) void k_ln_shift(const float* __restrict__ x,
        const float* __restrict__ w, const float* __restrict__ b,
        _Float16* __restrict__ xm) {
    int row = blockIdx.x;            // row = b*T + t
    int t = row % Tq;
    const float* src = x + (size_t)row * Dq;
    int d0 = threadIdx.x, d1 = threadIdx.x + 256;
    float a0 = src[d0], a1 = src[d1];
    float p0 = 0.f, p1 = 0.f;
    if (t > 0) { p0 = src[d0 - Dq]; p1 = src[d1 - Dq]; }

    float s  = a0 + a1, q  = a0 * a0 + a1 * a1;
    float sp = p0 + p1, qp = p0 * p0 + p1 * p1;
    __shared__ float sm[32];
    for (int o = 16; o > 0; o >>= 1) {
        s  += __shfl_down(s,  o, 32);  q  += __shfl_down(q,  o, 32);
        sp += __shfl_down(sp, o, 32);  qp += __shfl_down(qp, o, 32);
    }
    int lane = threadIdx.x & 31, wid = threadIdx.x >> 5;
    if (lane == 0) { sm[wid] = s; sm[wid + 8] = q; sm[wid + 16] = sp; sm[wid + 24] = qp; }
    __syncthreads();
    if (threadIdx.x == 0) {
        float t0 = 0.f, t1 = 0.f, t2 = 0.f, t3 = 0.f;
        for (int i = 0; i < 8; i++) {
            t0 += sm[i]; t1 += sm[i + 8]; t2 += sm[i + 16]; t3 += sm[i + 24];
        }
        sm[0] = t0; sm[8] = t1; sm[16] = t2; sm[24] = t3;
    }
    __syncthreads();
    float m    = sm[0]  * (1.0f / Dq);
    float var  = sm[8]  * (1.0f / Dq) - m * m;
    float inv  = rsqrtf(var + 1e-5f);
    float mp   = sm[16] * (1.0f / Dq);
    float varp = sm[24] * (1.0f / Dq) - mp * mp;
    float invp = rsqrtf(varp + 1e-5f);

    float n0 = (a0 - m) * inv * w[d0] + b[d0];
    float n1 = (a1 - m) * inv * w[d1] + b[d1];
    float pn0 = 0.f, pn1 = 0.f;
    if (t > 0) {                       // uniform branch per block
        pn0 = (p0 - mp) * invp * w[d0] + b[d0];
        pn1 = (p1 - mp) * invp * w[d1] + b[d1];
    }
    size_t base = (size_t)row * Dq;
    xm[base + d0] = (_Float16)(0.5f * (n0 + pn0));
    xm[base + d1] = (_Float16)(0.5f * (n1 + pn1));
}

// ---------------------------------------------------------------------------
// WMMA GEMM: C[M,N] = A[M,K] * W[N,K]^T (+epilogue)
// 256 threads = 8 waves (4x2); block tile 128x128; wave tile 32x64.
// MODE 0: f32 out, no bias.
// MODE 1: f32 out + bias.
// MODE 2: relu(A*W^T + bias)^2 -> f16 out.
// MODE 3: f32 out = kk*vv + aa*val  (kvab fusion, no bias).
// ---------------------------------------------------------------------------
template <int MODE>
__global__ __launch_bounds__(256) void k_wmma_gemm(
        const _Float16* __restrict__ A, const _Float16* __restrict__ W,
        const float* __restrict__ bias, float* __restrict__ Cf,
        _Float16* __restrict__ Ch,
        const float* __restrict__ kk, const float* __restrict__ vv,
        const float* __restrict__ aa,
        int Ndim, int Kdim) {
    int lane = threadIdx.x & 31;
    int wid  = threadIdx.x >> 5;
    int wm = wid & 3, wn = wid >> 2;               // 4 (M) x 2 (N) wave grid
    int m0 = blockIdx.x * 128 + wm * 32;
    int n0 = blockIdx.y * 128 + wn * 64;
    int lr = lane & 15;
    int hi = lane >> 4;                            // lane half

    v8f acc[2][4];
#pragma unroll
    for (int i = 0; i < 2; i++)
#pragma unroll
        for (int j = 0; j < 4; j++)
#pragma unroll
            for (int e = 0; e < 8; e++) acc[i][j][e] = 0.0f;

    // A fragment: lane lr holds row m0+i*16+lr; lo half K=k0..k0+15,
    // hi half K=k0+16..k0+31 (contiguous, matching B).
    const _Float16* arow[2];
#pragma unroll
    for (int i = 0; i < 2; i++)
        arow[i] = A + (size_t)(m0 + i * 16 + lr) * Kdim + hi * 16;
    // B fragment: column n of B == row n of W (both contiguous along K);
    // lo half K=k0..k0+15, hi half K=k0+16..k0+31 (16 contiguous halves).
    const _Float16* brow[4];
#pragma unroll
    for (int j = 0; j < 4; j++)
        brow[j] = W + (size_t)(n0 + j * 16 + lr) * Kdim + hi * 16;

    for (int k0 = 0; k0 < Kdim; k0 += 32) {
        v16h af[2], bf[4];
#pragma unroll
        for (int i = 0; i < 2; i++) {
            v8h c1 = *(const v8h*)(arow[i] + k0);
            v8h c2 = *(const v8h*)(arow[i] + k0 + 8);
            af[i] = __builtin_shufflevector(c1, c2, 0, 1, 2, 3, 4, 5, 6, 7,
                                                    8, 9, 10, 11, 12, 13, 14, 15);
        }
#pragma unroll
        for (int j = 0; j < 4; j++)
            bf[j] = *(const v16h*)(brow[j] + k0);
#pragma unroll
        for (int i = 0; i < 2; i++)
#pragma unroll
            for (int j = 0; j < 4; j++)
                acc[i][j] = __builtin_amdgcn_wmma_f32_16x16x32_f16(
                    false, af[i], false, bf[j], (short)0, acc[i][j], false, false);
    }

#pragma unroll
    for (int j = 0; j < 4; j++) {
        int ncol = n0 + j * 16 + lr;
        float bv = (MODE == 1 || MODE == 2) ? bias[ncol] : 0.0f;
#pragma unroll
        for (int i = 0; i < 2; i++) {
#pragma unroll
            for (int e = 0; e < 8; e++) {
                int row = m0 + i * 16 + hi * 8 + e;
                size_t idx = (size_t)row * Ndim + ncol;
                float val = acc[i][j][e] + bv;
                if (MODE == 2) {
                    val = fmaxf(val, 0.0f);
                    val = val * val;
                    Ch[idx] = (_Float16)val;
                } else if (MODE == 3) {
                    Cf[idx] = kk[idx] * vv[idx] + aa[idx] * val;
                } else {
                    Cf[idx] = val;
                }
            }
        }
    }
}

// ---------------------------------------------------------------------------
// Diagonal linear recurrence: s[t] = s[t-1]*w[d] + u[t]; 8192 channels
// ---------------------------------------------------------------------------
__global__ __launch_bounds__(256) void k_recur(const float* __restrict__ u,
        const float* __restrict__ decay, float* __restrict__ st,
        float* __restrict__ last) {
    int idx = blockIdx.x * 256 + threadIdx.x;      // 0 .. B*D-1
    int b = idx / Dq, d = idx % Dq;
    float w = __expf(-__expf(decay[d]));
    const float* src = u + (size_t)b * Tq * Dq + d;
    float* dst = st + (size_t)b * Tq * Dq + d;
    float s = 0.0f;
    float cur = src[0];
    for (int t = 0; t < Tq - 1; t++) {
        float nxt = src[(size_t)(t + 1) * Dq];     // prefetch next u
        s = s * w + cur;
        dst[(size_t)t * Dq] = s;
        cur = nxt;
    }
    s = s * w + cur;
    dst[(size_t)(Tq - 1) * Dq] = s;
    last[idx] = s;                                  // -> d_out states[l,b,d]
}

// ---------------------------------------------------------------------------
// GroupNorm (16 groups of 32 channels) + x += sigmoid(r)*ns*sigmoid(g)
// One block per row; each wave owns exactly groups (w) and (w+8).
// ---------------------------------------------------------------------------
__global__ __launch_bounds__(256) void k_gn_mix(const float* __restrict__ st,
        const float* __restrict__ sw, const float* __restrict__ sb,
        const float* __restrict__ r, const float* __restrict__ g,
        float* __restrict__ x) {
    size_t base = (size_t)blockIdx.x * Dq;
    int d0 = threadIdx.x, d1 = threadIdx.x + 256;
    float v0 = st[base + d0], v1 = st[base + d1];
    float s0 = v0, q0 = v0 * v0, s1 = v1, q1 = v1 * v1;
    for (int o = 16; o > 0; o >>= 1) {
        s0 += __shfl_xor(s0, o, 32); q0 += __shfl_xor(q0, o, 32);
        s1 += __shfl_xor(s1, o, 32); q1 += __shfl_xor(q1, o, 32);
    }
    float m0 = s0 * (1.0f / 32), var0 = q0 * (1.0f / 32) - m0 * m0;
    float m1 = s1 * (1.0f / 32), var1 = q1 * (1.0f / 32) - m1 * m1;
    float n0 = (v0 - m0) * rsqrtf(var0 + 1e-5f) * sw[d0] + sb[d0];
    float n1 = (v1 - m1) * rsqrtf(var1 + 1e-5f) * sw[d1] + sb[d1];
    x[base + d0] += sigf(r[base + d0]) * n0 * sigf(g[base + d0]);
    x[base + d1] += sigf(r[base + d1]) * n1 * sigf(g[base + d1]);
}

// ---------------------------------------------------------------------------
// GRN: gx[b,d] = sqrt(sum_t h^2)
// ---------------------------------------------------------------------------
__global__ __launch_bounds__(256) void k_grn_gx(const float* __restrict__ h,
                                                float* __restrict__ gx) {
    int idx = blockIdx.x * 256 + threadIdx.x;      // 0 .. B*D-1
    int b = idx / Dq, d = idx % Dq;
    const float* src = h + (size_t)b * Tq * Dq + d;
    float s = 0.0f;
    for (int t = 0; t < Tq; t++) {
        float v = src[(size_t)t * Dq];
        s += v * v;
    }
    gx[idx] = sqrtf(s);
}

// GRN: mean over channels per batch
__global__ __launch_bounds__(256) void k_grn_mean(const float* __restrict__ gx,
                                                  float* __restrict__ mv) {
    int b = blockIdx.x;
    float v = gx[b * Dq + threadIdx.x] + gx[b * Dq + threadIdx.x + 256];
    __shared__ float sm[8];
    for (int o = 16; o > 0; o >>= 1) v += __shfl_down(v, o, 32);
    int lane = threadIdx.x & 31, wid = threadIdx.x >> 5;
    if (lane == 0) sm[wid] = v;
    __syncthreads();
    if (threadIdx.x == 0) {
        float t = 0.f;
        for (int i = 0; i < 8; i++) t += sm[i];
        mv[b] = t * (1.0f / Dq);
    }
}

// GRN apply + final FFN residual: x += sigmoid(rr)*(gg*(h*nx)+gb+h)
__global__ void k_grn_apply(const float* __restrict__ h, const float* __restrict__ rr,
        const float* __restrict__ gx, const float* __restrict__ mv,
        const float* __restrict__ gg, const float* __restrict__ gb,
        float* __restrict__ x) {
    size_t i = (size_t)blockIdx.x * blockDim.x + threadIdx.x;
    if (i >= (size_t)Mq * Dq) return;
    int d = (int)(i % Dq);
    size_t bt = i / Dq;
    int b = (int)(bt / Tq);
    float hv = h[i];
    float nx = gx[(size_t)b * Dq + d] / (mv[b] + 1e-6f);
    float h2 = gg[d] * (hv * nx) + gb[d] + hv;
    x[i] += sigf(rr[i]) * h2;
}

// ---------------------------------------------------------------------------
// Host driver
// ---------------------------------------------------------------------------
extern "C" void kernel_launch(void* const* d_in, const int* in_sizes, int n_in,
                              void* d_out, int out_size, void* d_ws, size_t ws_size,
                              hipStream_t stream) {
    (void)in_sizes; (void)n_in; (void)out_size; (void)ws_size;
    const int*   y       = (const int*)  d_in[0];
    const float* embed   = (const float*)d_in[1];
    const float* ln_in_w = (const float*)d_in[2];
    const float* ln_in_b = (const float*)d_in[3];
    const float* ln1_w   = (const float*)d_in[4];
    const float* ln1_b   = (const float*)d_in[5];
    const float* ln2_w   = (const float*)d_in[6];
    const float* ln2_b   = (const float*)d_in[7];
    const float* sn_w    = (const float*)d_in[8];
    const float* sn_b    = (const float*)d_in[9];
    const float* grn_g   = (const float*)d_in[10];
    const float* grn_b   = (const float*)d_in[11];
    const float* Wr      = (const float*)d_in[12];
    const float* Wk      = (const float*)d_in[13];
    const float* Wv      = (const float*)d_in[14];
    const float* Wg      = (const float*)d_in[15];
    const float* Wa      = (const float*)d_in[16];
    const float* Wb      = (const float*)d_in[17];
    const float* decay   = (const float*)d_in[18];
    const float* Wffk    = (const float*)d_in[19];
    const float* bffk    = (const float*)d_in[20];
    const float* Wffv    = (const float*)d_in[21];
    const float* bffv    = (const float*)d_in[22];
    const float* Wffr    = (const float*)d_in[23];
    const float* bffr    = (const float*)d_in[24];

    char* ws = (char*)d_ws;
    size_t off = 0;
    auto alloc = [&](size_t bytes) -> void* {
        void* p = (void*)(ws + off);
        off += (bytes + 255) & ~(size_t)255;
        return p;
    };

    const size_t MD = (size_t)Mq * Dq;
    float*    x    = (float*)   alloc(MD * 4);
    _Float16* xmh  = (_Float16*)alloc(MD * 2);      // xm / xx2 (f16 GEMM input)
    float*    buf0 = (float*)   alloc(MD * 4);      // r
    float*    buf1 = (float*)   alloc(MD * 4);      // k          -> h
    float*    buf2 = (float*)   alloc(MD * 4);      // v          -> states
    float*    buf3 = (float*)   alloc(MD * 4);      // g          -> rr
    float*    buf4 = (float*)   alloc(MD * 4);      // a          -> kf f16 (M*F*2 == M*D*4)
    float*    buf5 = (float*)   alloc(MD * 4);      // kvab (fused bb GEMM)
    const size_t dd = (size_t)Dq * Dq, fd = (size_t)Fq * Dq;
    _Float16* whr  = (_Float16*)alloc((size_t)Lq * dd * 2);
    _Float16* whk  = (_Float16*)alloc((size_t)Lq * dd * 2);
    _Float16* whv  = (_Float16*)alloc((size_t)Lq * dd * 2);
    _Float16* whg  = (_Float16*)alloc((size_t)Lq * dd * 2);
    _Float16* wha  = (_Float16*)alloc((size_t)Lq * dd * 2);
    _Float16* whb  = (_Float16*)alloc((size_t)Lq * dd * 2);
    _Float16* whfk = (_Float16*)alloc((size_t)Lq * fd * 2);
    _Float16* whfv = (_Float16*)alloc((size_t)Lq * fd * 2);
    _Float16* whfr = (_Float16*)alloc((size_t)Lq * dd * 2);
    float*    gx   = (float*)   alloc((size_t)Bq * Dq * 4);
    float*    mv   = (float*)   alloc((size_t)Bq * 4);

    // Convert all weights fp32 -> f16 (deterministic, redone each call)
    auto conv = [&](const float* s, _Float16* d, size_t n) {
        k_f32_to_f16<<<2048, 256, 0, stream>>>(s, d, n);
    };
    conv(Wr, whr, (size_t)Lq * dd);  conv(Wk, whk, (size_t)Lq * dd);
    conv(Wv, whv, (size_t)Lq * dd);  conv(Wg, whg, (size_t)Lq * dd);
    conv(Wa, wha, (size_t)Lq * dd);  conv(Wb, whb, (size_t)Lq * dd);
    conv(Wffk, whfk, (size_t)Lq * fd);
    conv(Wffv, whfv, (size_t)Lq * fd);
    conv(Wffr, whfr, (size_t)Lq * dd);

    k_embed_ln<<<Mq, 256, 0, stream>>>(y, embed, ln_in_w, ln_in_b, x);

    dim3 g512(Mq / 128, Dq / 128);    // N = 512
    dim3 g1024(Mq / 128, Fq / 128);   // N = 1024
    int elemBlocks = (int)(MD / 256);
    float* last_base = (float*)d_out + MD;   // states tail: [L, B, D]

    for (int l = 0; l < Lq; l++) {
        // ---- time-mix half ----
        k_ln_shift<<<Mq, 256, 0, stream>>>(x, ln1_w + l * Dq, ln1_b + l * Dq, xmh);
        k_wmma_gemm<0><<<g512, 256, 0, stream>>>(xmh, whr + l * dd, nullptr, buf0, nullptr,
                                                 nullptr, nullptr, nullptr, Dq, Dq); // r
        k_wmma_gemm<0><<<g512, 256, 0, stream>>>(xmh, whk + l * dd, nullptr, buf1, nullptr,
                                                 nullptr, nullptr, nullptr, Dq, Dq); // k
        k_wmma_gemm<0><<<g512, 256, 0, stream>>>(xmh, whv + l * dd, nullptr, buf2, nullptr,
                                                 nullptr, nullptr, nullptr, Dq, Dq); // v
        k_wmma_gemm<0><<<g512, 256, 0, stream>>>(xmh, whg + l * dd, nullptr, buf3, nullptr,
                                                 nullptr, nullptr, nullptr, Dq, Dq); // g
        k_wmma_gemm<0><<<g512, 256, 0, stream>>>(xmh, wha + l * dd, nullptr, buf4, nullptr,
                                                 nullptr, nullptr, nullptr, Dq, Dq); // a
        // bb GEMM with fused kvab = k*v + a*bb epilogue
        k_wmma_gemm<3><<<g512, 256, 0, stream>>>(xmh, whb + l * dd, nullptr, buf5, nullptr,
                                                 buf1, buf2, buf4, Dq, Dq);          // kvab
        k_recur<<<(Bq * Dq) / 256, 256, 0, stream>>>(buf5, decay + l * Dq, buf2,
                                                     last_base + (size_t)l * Bq * Dq);
        k_gn_mix<<<Mq, 256, 0, stream>>>(buf2, sn_w + l * Dq, sn_b + l * Dq, buf0, buf3, x);

        // ---- channel-mix (FFN) half ----
        k_ln_f16<<<Mq, 256, 0, stream>>>(x, ln2_w + l * Dq, ln2_b + l * Dq, xmh);
        k_wmma_gemm<2><<<g1024, 256, 0, stream>>>(xmh, whfk + l * fd, bffk + l * Fq,
                                                  nullptr, (_Float16*)buf4,
                                                  nullptr, nullptr, nullptr, Fq, Dq); // kf = relu^2
        k_wmma_gemm<1><<<g512, 256, 0, stream>>>((const _Float16*)buf4, whfv + l * fd,
                                                 bffv + l * Dq, buf1, nullptr,
                                                 nullptr, nullptr, nullptr, Dq, Fq);  // h
        k_wmma_gemm<1><<<g512, 256, 0, stream>>>(xmh, whfr + l * dd, bffr + l * Dq,
                                                 buf3, nullptr,
                                                 nullptr, nullptr, nullptr, Dq, Dq);  // rr
        k_grn_gx<<<(Bq * Dq) / 256, 256, 0, stream>>>(buf1, gx);
        k_grn_mean<<<Bq, 256, 0, stream>>>(gx, mv);
        k_grn_apply<<<elemBlocks, 256, 0, stream>>>(buf1, buf3, gx, mv,
                                                    grn_g + l * Dq, grn_b + l * Dq, x);
    }

    hipMemcpyAsync(d_out, x, MD * 4, hipMemcpyDeviceToDevice, stream);
}
